// QwenA1_31533649887534
// MI455X (gfx1250) — matrix operand order repaired and checked by reading, not compiled, not run on hardware
//
#include <hip/hip_runtime.h>
#include <hip/hip_bf16.h>

#define S_TOT_C 1344
#define NHEADS_C 16
#define NKV_C 8
#define HD_C 128
#define BATCH_C 2

typedef __bf16 bf16;
typedef __attribute__((ext_vector_type(16))) __bf16 v16bf;
typedef __attribute__((ext_vector_type(8)))  float  v8f;
typedef __attribute__((ext_vector_type(4)))  int    v4i;

#if defined(__AMDGCN__) && __has_builtin(__builtin_amdgcn_global_load_async_to_lds_b128) && __has_builtin(__builtin_amdgcn_s_wait_asynccnt)
#define USE_ASYNC_LDS 1
typedef __attribute__((address_space(1))) v4i gv4i;
typedef __attribute__((address_space(3))) v4i lv4i;
#else
#define USE_ASYNC_LDS 0
#endif

union Frag { v16bf v; uint4 q[2]; };

__device__ __forceinline__ v8f wmma_bf16(v16bf a, v16bf b, v8f c) {
  // emits v_wmma_f32_16x16x32_bf16
  return __builtin_amdgcn_wmma_f32_16x16x32_bf16(false, a, false, b, (short)0, c, false, false);
}

// ---------------------------------------------------------------------------
// f32 -> bf16 conversion (grid-stride)
// ---------------------------------------------------------------------------
__global__ void k_f32_to_bf16(const float* __restrict__ src, bf16* __restrict__ dst, long n) {
  long i = (long)blockIdx.x * blockDim.x + threadIdx.x;
  long stride = (long)gridDim.x * blockDim.x;
  for (; i < n; i += stride) dst[i] = (bf16)src[i];
}

// ---------------------------------------------------------------------------
// RMSNorm over rows -> bf16   (one block per row)
// ---------------------------------------------------------------------------
__global__ __launch_bounds__(256)
void k_rmsnorm_bf16(const float* __restrict__ x, const float* __restrict__ w,
                    bf16* __restrict__ out, int cols) {
  long row = blockIdx.x;
  const float* xr = x + row * (long)cols;
  float ss = 0.0f;
  for (int c = threadIdx.x; c < cols; c += 256) { float v = xr[c]; ss += v * v; }
  __shared__ float red[256];
  red[threadIdx.x] = ss;
  __syncthreads();
  for (int s = 128; s > 0; s >>= 1) {
    if (threadIdx.x < s) red[threadIdx.x] += red[threadIdx.x + s];
    __syncthreads();
  }
  float rs = rsqrtf(red[0] / (float)cols + 1e-6f);
  bf16* orow = out + row * (long)cols;
  for (int c = threadIdx.x; c < cols; c += 256) orow[c] = (bf16)(xr[c] * rs * w[c]);
}

// ---------------------------------------------------------------------------
// Generic batched WMMA GEMM:  C[z] = alpha * A[z] @ W[z]^T + addmat[z]
//   A: [M,K] bf16 row-major, per-z offset z*zsA
//   W: [N,K] bf16 row-major, per-z offset (z/zdW)*zsW
//   C: [M,N] f32,            per-z offset z*zsC
//   addmat (optional f32 [M,N]): per-z offset (z/zdAdd)*zsAdd
// Double-buffered LDS; global->LDS via async-to-LDS DMA when available.
// All of M%BM, N%BN, K%BK must be 0 (guaranteed by problem shapes).
// ---------------------------------------------------------------------------
template<int BM, int BN, int BK, int WM, int WN>
__global__ __launch_bounds__(256)
void k_gemm(const bf16* __restrict__ A, const bf16* __restrict__ W, float* __restrict__ C,
            int M, int N, int K,
            long zsA, int zdW, long zsW, long zsC,
            float alpha, const float* __restrict__ addm, int zdAdd, long zsAdd) {
  constexpr int WAVES_N = BN / WN;
  constexpr int MF = WM / 16;
  constexpr int NF = WN / 16;
  constexpr int CPR = BK / 8;                       // 16B chunks per tile row
  constexpr int TCH = (BM + BN) * CPR / 256;        // async chunks per thread per slab

  const int z = blockIdx.z;
  const bf16* Ab = A + (long)z * zsA + (long)blockIdx.y * BM * K;
  const bf16* Wb = W + (long)(z / zdW) * zsW + (long)blockIdx.x * BN * K;
  float* Cb = C + (long)z * zsC;
  const float* addb = addm ? (addm + (long)(z / zdAdd) * zsAdd) : nullptr;

  __shared__ alignas(16) bf16 As[2][BM * BK];
  __shared__ alignas(16) bf16 Ws[2][BN * BK];

  const int tid  = threadIdx.x;
  const int wave = tid >> 5;
  const int lane = tid & 31;
  const int wn   = wave % WAVES_N;
  const int wm   = wave / WAVES_N;
  const int half = lane >> 4;
  const int r    = lane & 15;

  v8f acc[MF][NF] = {};

  // stage one K-slab (k0) of A and W tiles into LDS buffer `nbuf`
  auto issue = [&](int nbuf, int k0) {
    for (int it = tid; it < BM * CPR; it += 256) {
      int rr = it / CPR, cc = (it % CPR) * 8;
#if USE_ASYNC_LDS
      __builtin_amdgcn_global_load_async_to_lds_b128(
          (gv4i*)&Ab[(long)rr * K + k0 + cc],
          (lv4i*)&As[nbuf][rr * BK + cc], 0, 0);
#else
      *(uint4*)&As[nbuf][rr * BK + cc] = *(const uint4*)&Ab[(long)rr * K + k0 + cc];
#endif
    }
    for (int it = tid; it < BN * CPR; it += 256) {
      int rr = it / CPR, cc = (it % CPR) * 8;
#if USE_ASYNC_LDS
      __builtin_amdgcn_global_load_async_to_lds_b128(
          (gv4i*)&Wb[(long)rr * K + k0 + cc],
          (lv4i*)&Ws[nbuf][rr * BK + cc], 0, 0);
#else
      *(uint4*)&Ws[nbuf][rr * BK + cc] = *(const uint4*)&Wb[(long)rr * K + k0 + cc];
#endif
    }
  };

  issue(0, 0);
  int buf = 0;

  for (int k0 = 0; k0 < K; k0 += BK) {
    if (k0 + BK < K) {
      issue(buf ^ 1, k0 + BK);          // overlap next slab's DMA with this slab's math
#if USE_ASYNC_LDS
      __builtin_amdgcn_s_wait_asynccnt(TCH);   // allow newest batch in flight; older landed
#endif
    } else {
#if USE_ASYNC_LDS
      __builtin_amdgcn_s_wait_asynccnt(0);
#endif
    }
    __syncthreads();

#pragma unroll
    for (int kk = 0; kk < BK; kk += 32) {
      v16bf af[MF], bfrag[NF];
#pragma unroll
      for (int mi = 0; mi < MF; ++mi) {
        // A 16x32 fragment: lane<16 -> K 0..7 & 16..23 ; lane>=16 -> K 8..15 & 24..31
        const bf16* p = &As[buf][(wm * WM + mi * 16 + r) * BK + kk];
        Frag u;
        u.q[0] = *(const uint4*)(p + half * 8);
        u.q[1] = *(const uint4*)(p + 16 + half * 8);
        af[mi] = u.v;
      }
#pragma unroll
      for (int ni = 0; ni < NF; ++ni) {
        // B 32x16 fragment: lane<16 -> K 0..7 & 16..23 of col n ; lane>=16 -> K 8..15 & 24..31
        const bf16* p = &Ws[buf][(wn * WN + ni * 16 + r) * BK + kk];
        Frag u;
        u.q[0] = *(const uint4*)(p + half * 8);
        u.q[1] = *(const uint4*)(p + 16 + half * 8);
        bfrag[ni] = u.v;
      }
#pragma unroll
      for (int mi = 0; mi < MF; ++mi)
#pragma unroll
        for (int ni = 0; ni < NF; ++ni)
          acc[mi][ni] = wmma_bf16(af[mi], bfrag[ni], acc[mi][ni]);
    }
    __syncthreads();
    buf ^= 1;
  }

  // C/D layout: VGPR i -> row (half*8 + i), col = lane&15
#pragma unroll
  for (int mi = 0; mi < MF; ++mi) {
    int rowb = blockIdx.y * BM + wm * WM + mi * 16 + half * 8;
#pragma unroll
    for (int ni = 0; ni < NF; ++ni) {
      int col = blockIdx.x * BN + wn * WN + ni * 16 + r;
#pragma unroll
      for (int i = 0; i < 8; ++i) {
        long idx = (long)(rowb + i) * N + col;
        float v = alpha * acc[mi][ni][i];
        if (addb) v += addb[idx];
        Cb[idx] = v;
      }
    }
  }
}

// ---------------------------------------------------------------------------
// Per-head q/k RMSNorm + RoPE + scatter into global Q/K (row-major) and
// V transposed [b, hk, d, s].  grid = (B*S_e, 24), block = 128 (one per d).
//   blockIdx.y < 16 : q head ; >= 16 : kv head (handles both k and v)
// ---------------------------------------------------------------------------
__global__ __launch_bounds__(128)
void k_qknorm_rope(const float* __restrict__ qbuf, const float* __restrict__ kbuf,
                   const float* __restrict__ vbuf,
                   const float* __restrict__ qn, const float* __restrict__ kn,
                   const int* __restrict__ posids,
                   bf16* __restrict__ Qg, bf16* __restrict__ Kg, bf16* __restrict__ Vt,
                   int S_e, int seg) {
  const int row = blockIdx.x;         // b * S_e + s
  const int hh  = blockIdx.y;
  const int d   = threadIdx.x;
  const int b   = row / S_e;
  const int s   = row % S_e;
  const int gs  = seg + s;
  const int pos = posids[b * S_TOT_C + gs];

  const int fi = d & 63;
  const float invf = powf(5000000.0f, -((float)fi) / 64.0f);
  const float ang = (float)pos * invf;
  const float cs = cosf(ang);
  const float sn = sinf(ang);

  __shared__ float sh[128];
  __shared__ float red[128];

  if (hh < NHEADS_C) {
    const float v = qbuf[(long)row * (NHEADS_C * HD_C) + hh * HD_C + d];
    red[d] = v * v;
    __syncthreads();
    for (int st = 64; st > 0; st >>= 1) { if (d < st) red[d] += red[d + st]; __syncthreads(); }
    const float rs = rsqrtf(red[0] / 128.0f + 1e-6f);
    const float nv = v * rs * qn[d];
    sh[d] = nv;
    __syncthreads();
    const float partner = sh[d ^ 64];
    const float o = (d < 64) ? (nv * cs - partner * sn) : (nv * cs + partner * sn);
    Qg[(((long)(b * NHEADS_C + hh)) * S_TOT_C + gs) * HD_C + d] = (bf16)o;
  } else {
    const int h = hh - NHEADS_C;
    const float v = kbuf[(long)row * (NKV_C * HD_C) + h * HD_C + d];
    red[d] = v * v;
    __syncthreads();
    for (int st = 64; st > 0; st >>= 1) { if (d < st) red[d] += red[d + st]; __syncthreads(); }
    const float rs = rsqrtf(red[0] / 128.0f + 1e-6f);
    const float nv = v * rs * kn[d];
    sh[d] = nv;
    __syncthreads();
    const float partner = sh[d ^ 64];
    const float o = (d < 64) ? (nv * cs - partner * sn) : (nv * cs + partner * sn);
    Kg[(((long)(b * NKV_C + h)) * S_TOT_C + gs) * HD_C + d] = (bf16)o;
    const float vv = vbuf[(long)row * (NKV_C * HD_C) + h * HD_C + d];
    Vt[(((long)(b * NKV_C + h)) * HD_C + d) * S_TOT_C + gs] = (bf16)vv;
  }
}

// ---------------------------------------------------------------------------
// Row softmax (length n) f32 -> bf16. One block per row.
// ---------------------------------------------------------------------------
__global__ __launch_bounds__(256)
void k_softmax(const float* __restrict__ S, bf16* __restrict__ P, int n) {
  long row = blockIdx.x;
  const float* sr = S + row * (long)n;
  __shared__ float red[256];
  float mx = -3.4e38f;
  for (int c = threadIdx.x; c < n; c += 256) mx = fmaxf(mx, sr[c]);
  red[threadIdx.x] = mx;
  __syncthreads();
  for (int s = 128; s > 0; s >>= 1) {
    if (threadIdx.x < s) red[threadIdx.x] = fmaxf(red[threadIdx.x], red[threadIdx.x + s]);
    __syncthreads();
  }
  mx = red[0];
  __syncthreads();
  float sum = 0.0f;
  for (int c = threadIdx.x; c < n; c += 256) sum += expf(sr[c] - mx);
  red[threadIdx.x] = sum;
  __syncthreads();
  for (int s = 128; s > 0; s >>= 1) {
    if (threadIdx.x < s) red[threadIdx.x] += red[threadIdx.x + s];
    __syncthreads();
  }
  const float inv = 1.0f / red[0];
  bf16* pr = P + row * (long)n;
  for (int c = threadIdx.x; c < n; c += 256) pr[c] = (bf16)(expf(sr[c] - mx) * inv);
}

// ---------------------------------------------------------------------------
// [B,H,S,D] f32 -> [B,S,H*D] bf16 (output-linear for coalesced writes)
// ---------------------------------------------------------------------------
__global__ void k_att_reorder(const float* __restrict__ src, bf16* __restrict__ dst) {
  const long n = (long)BATCH_C * S_TOT_C * NHEADS_C * HD_C;
  long i = (long)blockIdx.x * blockDim.x + threadIdx.x;
  long stride = (long)gridDim.x * blockDim.x;
  for (; i < n; i += stride) {
    long d = i & (HD_C - 1);
    long h = (i >> 7) & (NHEADS_C - 1);
    long q = (i >> 11) % S_TOT_C;
    long b = i / ((long)S_TOT_C * NHEADS_C * HD_C);
    dst[i] = (bf16)src[(((b * NHEADS_C + h) * S_TOT_C + q) << 7) + d];
  }
}

// ---------------------------------------------------------------------------
// act = silu(g) * u  -> bf16
// ---------------------------------------------------------------------------
__global__ void k_silu_mul(const float* __restrict__ g, const float* __restrict__ u,
                           bf16* __restrict__ out, long n) {
  long i = (long)blockIdx.x * blockDim.x + threadIdx.x;
  long stride = (long)gridDim.x * blockDim.x;
  for (; i < n; i += stride) {
    float x = g[i];
    out[i] = (bf16)((x / (1.0f + expf(-x))) * u[i]);
  }
}

// ---------------------------------------------------------------------------
// Host orchestration
// ---------------------------------------------------------------------------
extern "C" void kernel_launch(void* const* d_in, const int* in_sizes, int n_in,
                              void* d_out, int out_size, void* d_ws, size_t ws_size,
                              hipStream_t stream) {
  (void)in_sizes; (void)n_in; (void)out_size; (void)ws_size;

  struct EInfo { int hidden, inter, S, seg; long outoff; };
  const EInfo ex[3] = {
    {2048, 6144, 1024, 0,    0},
    {1024, 3072, 256,  1024, (long)BATCH_C * 1024 * 2048},
    {1024, 3072, 64,   1280, (long)BATCH_C * 1024 * 2048 + (long)BATCH_C * 256 * 1024},
  };

  const float *xin[3], *ln[3], *wq[3], *wk[3], *wv[3], *qn[3], *kn[3], *wo[3], *pln[3], *wg[3], *wu[3], *wd[3];
  for (int e = 0; e < 3; ++e) {
    xin[e] = (const float*)d_in[e];
    int p = 3 + e * 11;
    ln[e]  = (const float*)d_in[p + 0];
    wq[e]  = (const float*)d_in[p + 1];
    wk[e]  = (const float*)d_in[p + 2];
    wv[e]  = (const float*)d_in[p + 3];
    qn[e]  = (const float*)d_in[p + 4];
    kn[e]  = (const float*)d_in[p + 5];
    wo[e]  = (const float*)d_in[p + 6];
    pln[e] = (const float*)d_in[p + 7];
    wg[e]  = (const float*)d_in[p + 8];
    wu[e]  = (const float*)d_in[p + 9];
    wd[e]  = (const float*)d_in[p + 10];
  }
  const float* mask = (const float*)d_in[36];
  const int*   pos  = (const int*)d_in[37];
  float* out = (float*)d_out;

  char* base = (char*)d_ws;
  size_t off = 0;
  auto take = [&](size_t bytes) -> void* {
    size_t a = (off + 255) & ~(size_t)255;
    off = a + bytes;
    return (void*)(base + a);
  };

  auto conv = [&](const float* s, bf16* d, long n) {
    long blocks = (n + 2047) / 2048; if (blocks > 8192) blocks = 8192; if (blocks < 1) blocks = 1;
    k_f32_to_bf16<<<dim3((unsigned)blocks), dim3(256), 0, stream>>>(s, d, n);
  };

  // ---- workspace layout ------------------------------------------------
  bf16 *wqb[3], *wkb[3], *wvb[3], *wob[3], *wgb[3], *wub[3], *wdb[3];
  bf16 *h[3], *h2[3], *actb[3];
  float *qf[3], *kf[3], *vf[3], *ob[3], *gf[3], *uf[3];
  for (int e = 0; e < 3; ++e) {
    const long hid = ex[e].hidden, itr = ex[e].inter, rows = (long)BATCH_C * ex[e].S;
    wqb[e]  = (bf16*)take(sizeof(bf16) * 2048 * hid);
    wkb[e]  = (bf16*)take(sizeof(bf16) * 1024 * hid);
    wvb[e]  = (bf16*)take(sizeof(bf16) * 1024 * hid);
    wob[e]  = (bf16*)take(sizeof(bf16) * hid * 2048);
    wgb[e]  = (bf16*)take(sizeof(bf16) * itr * hid);
    wub[e]  = (bf16*)take(sizeof(bf16) * itr * hid);
    wdb[e]  = (bf16*)take(sizeof(bf16) * hid * itr);
    h[e]    = (bf16*)take(sizeof(bf16) * rows * hid);
    h2[e]   = (bf16*)take(sizeof(bf16) * rows * hid);
    actb[e] = (bf16*)take(sizeof(bf16) * rows * itr);
    qf[e]   = (float*)take(sizeof(float) * rows * 2048);
    kf[e]   = (float*)take(sizeof(float) * rows * 1024);
    vf[e]   = (float*)take(sizeof(float) * rows * 1024);
    ob[e]   = (float*)take(sizeof(float) * rows * hid);
    gf[e]   = (float*)take(sizeof(float) * rows * itr);
    uf[e]   = (float*)take(sizeof(float) * rows * itr);
  }
  const long QELEM = (long)BATCH_C * NHEADS_C * S_TOT_C * HD_C;
  const long KELEM = (long)BATCH_C * NKV_C * S_TOT_C * HD_C;
  const long SCELEM = (long)BATCH_C * NHEADS_C * S_TOT_C * S_TOT_C;
  bf16*  Qg  = (bf16*)take(sizeof(bf16) * QELEM);
  bf16*  Kg  = (bf16*)take(sizeof(bf16) * KELEM);
  bf16*  Vt  = (bf16*)take(sizeof(bf16) * KELEM);
  float* sc  = (float*)take(sizeof(float) * SCELEM);
  bf16*  P   = (bf16*)take(sizeof(bf16) * SCELEM);
  float* ao  = (float*)take(sizeof(float) * QELEM);
  bf16*  att = (bf16*)take(sizeof(bf16) * (long)BATCH_C * S_TOT_C * 2048);

  // ---- 1) weights -> bf16 ---------------------------------------------
  for (int e = 0; e < 3; ++e) {
    const long hid = ex[e].hidden, itr = ex[e].inter;
    conv(wq[e], wqb[e], 2048 * hid);
    conv(wk[e], wkb[e], 1024 * hid);
    conv(wv[e], wvb[e], 1024 * hid);
    conv(wo[e], wob[e], hid * 2048);
    conv(wg[e], wgb[e], itr * hid);
    conv(wu[e], wub[e], itr * hid);
    conv(wd[e], wdb[e], hid * itr);
  }

  // ---- 2) pre-attention: RMSNorm, QKV GEMMs, qk-norm + RoPE -----------
  for (int e = 0; e < 3; ++e) {
    const int hid = ex[e].hidden;
    const int M = BATCH_C * ex[e].S;
    k_rmsnorm_bf16<<<dim3(M), dim3(256), 0, stream>>>(xin[e], ln[e], h[e], hid);

    k_gemm<128, 128, 64, 64, 32><<<dim3(2048 / 128, M / 128, 1), dim3(256), 0, stream>>>(
        h[e], wqb[e], qf[e], M, 2048, hid, 0, 1, 0, 0, 1.0f, nullptr, 1, 0);
    k_gemm<128, 128, 64, 64, 32><<<dim3(1024 / 128, M / 128, 1), dim3(256), 0, stream>>>(
        h[e], wkb[e], kf[e], M, 1024, hid, 0, 1, 0, 0, 1.0f, nullptr, 1, 0);
    k_gemm<128, 128, 64, 64, 32><<<dim3(1024 / 128, M / 128, 1), dim3(256), 0, stream>>>(
        h[e], wvb[e], vf[e], M, 1024, hid, 0, 1, 0, 0, 1.0f, nullptr, 1, 0);

    k_qknorm_rope<<<dim3(M, 24), dim3(128), 0, stream>>>(
        qf[e], kf[e], vf[e], qn[e], kn[e], pos, Qg, Kg, Vt, ex[e].S, ex[e].seg);
  }

  // ---- 3) attention ----------------------------------------------------
  const float scale = 0.08838834764831845f; // 1/sqrt(128)
  // scores[z] = scale * Q[z] @ K[z>>1]^T + mask[z/16]   (z = b*16 + h)
  k_gemm<64, 64, 64, 16, 32><<<dim3(S_TOT_C / 64, S_TOT_C / 64, BATCH_C * NHEADS_C), dim3(256), 0, stream>>>(
      Qg, Kg, sc, S_TOT_C, S_TOT_C, HD_C,
      (long)S_TOT_C * HD_C, 2, (long)S_TOT_C * HD_C, (long)S_TOT_C * S_TOT_C,
      scale, mask, 16, (long)S_TOT_C * S_TOT_C);

  k_softmax<<<dim3(BATCH_C * NHEADS_C * S_TOT_C), dim3(256), 0, stream>>>(sc, P, S_TOT_C);

  // out[z] = P[z] @ Vt[z>>1]^T  (Vt rows are head-dim d, length S)
  k_gemm<64, 128, 64, 32, 32><<<dim3(HD_C / 128, S_TOT_C / 64, BATCH_C * NHEADS_C), dim3(256), 0, stream>>>(
      P, Vt, ao, S_TOT_C, HD_C, S_TOT_C,
      (long)S_TOT_C * S_TOT_C, 2, (long)HD_C * S_TOT_C, (long)S_TOT_C * HD_C,
      1.0f, nullptr, 1, 0);

  k_att_reorder<<<dim3(4096), dim3(256), 0, stream>>>(ao, att);

  // ---- 4) post-attention per expert -----------------------------------
  for (int e = 0; e < 3; ++e) {
    const int hid = ex[e].hidden, itr = ex[e].inter, Se = ex[e].S;
    const int M = BATCH_C * Se;

    // o = att[seg] @ wo^T + x   (batched over b: att row stride S_TOT*2048)
    k_gemm<64, 128, 64, 32, 32><<<dim3(hid / 128, Se / 64, BATCH_C), dim3(256), 0, stream>>>(
        att + (long)ex[e].seg * 2048, wob[e], ob[e], Se, hid, 2048,
        (long)S_TOT_C * 2048, 1, 0, (long)Se * hid,
        1.0f, xin[e], 1, (long)Se * hid);

    k_rmsnorm_bf16<<<dim3(M), dim3(256), 0, stream>>>(ob[e], pln[e], h2[e], hid);

    k_gemm<128, 128, 64, 64, 32><<<dim3(itr / 128, M / 128, 1), dim3(256), 0, stream>>>(
        h2[e], wgb[e], gf[e], M, itr, hid, 0, 1, 0, 0, 1.0f, nullptr, 1, 0);
    k_gemm<128, 128, 64, 64, 32><<<dim3(itr / 128, M / 128, 1), dim3(256), 0, stream>>>(
        h2[e], wub[e], uf[e], M, itr, hid, 0, 1, 0, 0, 1.0f, nullptr, 1, 0);

    const long nact = (long)M * itr;
    long sb = (nact + 2047) / 2048; if (sb > 8192) sb = 8192;
    k_silu_mul<<<dim3((unsigned)sb), dim3(256), 0, stream>>>(gf[e], uf[e], actb[e], nact);

    // final: out = act @ wd^T + o
    k_gemm<128, 128, 64, 64, 32><<<dim3(hid / 128, M / 128, 1), dim3(256), 0, stream>>>(
        actb[e], wdb[e], out + ex[e].outoff, M, hid, itr, 0, 1, 0, 0,
        1.0f, ob[e], 1, 0);
  }
}
